// CrossAttention2d_49220325212799
// MI455X (gfx1250) — matrix operand order repaired and checked by reading, not compiled, not run on hardware
//
#include <hip/hip_runtime.h>
#include <hip/hip_bf16.h>

typedef __attribute__((ext_vector_type(16))) _Float16 v16h;
typedef __attribute__((ext_vector_type(8)))  float    v8f;

static_assert(sizeof(v16h) == 32, "v16h must be 32 bytes");

#define BATCH 32
#define CDIM  256
#define DDIM  128
#define NPIX  1024   // 32*32

__device__ __forceinline__ v8f wmma16(v16h a, v16h b, v8f c) {
    // D = A(16x32 f16) * B(32x16 f16) + C(16x16 f32)
    return __builtin_amdgcn_wmma_f32_16x16x32_f16(
        /*neg_a=*/false, a, /*neg_b=*/false, b,
        /*c_mod=*/(short)0, c, /*reuse_a=*/false, /*reuse_b=*/false);
}

__device__ __forceinline__ float redmax16(float v) {
    #pragma unroll
    for (int m = 1; m < 16; m <<= 1) v = fmaxf(v, __shfl_xor(v, m));
    return v;
}
__device__ __forceinline__ float redsum16(float v) {
    #pragma unroll
    for (int m = 1; m < 16; m <<= 1) v += __shfl_xor(v, m);
    return v;
}
__device__ __forceinline__ float redsum32(float v) {
    #pragma unroll
    for (int m = 1; m < 32; m <<= 1) v += __shfl_xor(v, m);
    return v;
}

union H16 { v16h v; uint4 u[2]; };

// Async copy of 16 bytes/lane from global to LDS (CDNA5 path, ASYNCcnt).
__device__ __forceinline__ void async_b128_to_lds(unsigned lds_byte_addr,
                                                  unsigned glb_byte_off,
                                                  const void* base) {
    asm volatile("global_load_async_to_lds_b128 %0, %1, %2"
                 :
                 : "v"(lds_byte_addr), "v"(glb_byte_off), "s"(base)
                 : "memory");
}
__device__ __forceinline__ void wait_asynccnt0() {
    asm volatile("s_wait_asynccnt 0" ::: "memory");
}

// ---------------------------------------------------------------------------
// Utility kernels
// ---------------------------------------------------------------------------
__global__ void zero_kernel(float* p, int n) {
    int i = blockIdx.x * 256 + threadIdx.x;
    if (i < n) p[i] = 0.0f;
}

__global__ void cvt_kernel(const float* __restrict__ src, _Float16* __restrict__ dst, int n) {
    int i = blockIdx.x * 256 + threadIdx.x;
    if (i < n) dst[i] = (_Float16)src[i];
}

// ---------------------------------------------------------------------------
// Projection: Y(N x O) = X(N x C) * W^T, X channel-major (B,C,N) f32,
// W f16 (O x C). Output f16 row-major (B,N,O) or transposed (B,O,N).
// One wave computes a 16-row x O tile. 4 waves / block, wave-private LDS slice.
// The X tile is staged into LDS with GLOBAL_LOAD_ASYNC_TO_LDS_B128 (f32),
// converted to f16 while building A-fragments.
// ---------------------------------------------------------------------------
template <int O, bool TR>
__global__ __launch_bounds__(128) void proj_kernel(const float* __restrict__ X,
                                                   const _Float16* __restrict__ Wh,
                                                   _Float16* __restrict__ Y) {
    __shared__ float lds[4][16 * CDIM];   // [wave][c*16 + m], 16KB per wave
    const int w    = threadIdx.x >> 5;
    const int lane = threadIdx.x & 31;
    const int lm   = lane & 15;
    const int lh   = lane >> 4;            // lane half
    const int b    = blockIdx.x >> 4;      // 16 blocks per batch
    const int n0   = (blockIdx.x & 15) * 64 + w * 16;
    const int abase = lh ? 8 : 0;          // A-fragment K base per lane half
    const int kbB   = lh ? 16 : 0;         // unused; B-fragment K base is abase too

    // Async-stage X tile (16 rows x 256 channels, f32) into the wave's slice.
    // Memory is contiguous along n for fixed c -> direct byte copy.
    {
        const unsigned lds0 = (unsigned)(size_t)(&lds[w][0]);
        #pragma unroll
        for (int cc = 0; cc < 8; ++cc) {
            int c = cc * 32 + lane;
            unsigned goff = (unsigned)((((size_t)b * CDIM + c) * NPIX + n0) * 4);
            unsigned loff = lds0 + (unsigned)(c * 16 * 4);
            #pragma unroll
            for (int i = 0; i < 4; ++i)
                async_b128_to_lds(loff + i * 16, goff + i * 16, X);
        }
        wait_asynccnt0();
    }
    // wave-private LDS slice -> no cross-wave barrier needed

    constexpr int NT = O / 16;
    v8f acc[NT];
    #pragma unroll
    for (int ot = 0; ot < NT; ++ot) acc[ot] = (v8f){0,0,0,0,0,0,0,0};

    #pragma unroll
    for (int kc = 0; kc < 8; ++kc) {
        v16h a;
        #pragma unroll
        for (int h = 0; h < 16; ++h) {
            int k = kc * 32 + ((h < 8) ? (abase + h) : (abase + 16 + (h - 8)));
            a[h] = (_Float16)lds[w][k * 16 + lm];
        }
        #pragma unroll
        for (int ot = 0; ot < NT; ++ot) {
            int o = ot * 16 + lm;
            H16 bu; bu.u[0] = *(const uint4*)(Wh + (size_t)o * CDIM + kc * 32 + abase); bu.u[1] = *(const uint4*)(Wh + (size_t)o * CDIM + kc * 32 + abase + 16); v16h bf = bu.v;
            acc[ot] = wmma16(a, bf, acc[ot]);
        }
    }

    #pragma unroll
    for (int ot = 0; ot < NT; ++ot) {
        int o = ot * 16 + lm;
        #pragma unroll
        for (int r = 0; r < 8; ++r) {
            int n = n0 + r + (lh ? 8 : 0);
            _Float16 hv = (_Float16)acc[ot][r];
            if (TR) Y[((size_t)b * O + o) * NPIX + n] = hv;
            else    Y[((size_t)b * NPIX + n) * O + o] = hv;
        }
    }
}

// ---------------------------------------------------------------------------
// Attention: Out(N x C) = softmax(Q K^T / sqrt(128)) V.
// Q,K f16 row-major (B,N,128); V f16 transposed (B,C,N); Out f16 (B,N,C).
// Wave handles 16 query rows x 128 channels (flash-style online softmax).
// Block = 4 waves: 2 row-tiles x 2 channel-halves.
// ---------------------------------------------------------------------------
__global__ __launch_bounds__(128) void attn_kernel(const _Float16* __restrict__ Q,
                                                   const _Float16* __restrict__ Km,
                                                   const _Float16* __restrict__ Vt,
                                                   _Float16* __restrict__ Out) {
    __shared__ _Float16 pbuf[4][32 * 16];   // wave-private P tile [key][row]
    const int w    = threadIdx.x >> 5;
    const int lane = threadIdx.x & 31;
    const int lm   = lane & 15;
    const int lh   = lane >> 4;
    const int rt   = w >> 1;                // row tile in block (0..1)
    const int ch   = w & 1;                 // channel half (0..1)
    const int b    = blockIdx.x >> 5;       // 32 blocks per batch
    const int n0   = (blockIdx.x & 31) * 32 + rt * 16;
    const int abase = lh ? 8 : 0;
    const int kbB   = lh ? 16 : 0;

    // Q fragments for the whole 16x128 row tile, kept in registers.
    H16 qf[4];
    {
        const _Float16* qrow = Q + ((size_t)b * NPIX + n0 + lm) * DDIM;
        #pragma unroll
        for (int kc = 0; kc < 4; ++kc) {
            qf[kc].u[0] = *(const uint4*)(qrow + kc * 32 + abase);
            qf[kc].u[1] = *(const uint4*)(qrow + kc * 32 + abase + 16);
        }
    }

    v8f acc[8];
    #pragma unroll
    for (int ot = 0; ot < 8; ++ot) acc[ot] = (v8f){0,0,0,0,0,0,0,0};
    float rmax[8], rsum[8];
    #pragma unroll
    for (int r = 0; r < 8; ++r) { rmax[r] = -1e30f; rsum[r] = 0.0f; }

    const float scale = 0.08838834764831845f;   // 1/sqrt(128)

    for (int j0 = 0; j0 < NPIX; j0 += 32) {
        // Prefetch next key/value block into caches (global_prefetch_b8).
        if (j0 + 32 < NPIX) {
            __builtin_prefetch((const void*)(Km + ((size_t)b * NPIX + j0 + 32 + lm) * DDIM), 0, 0);
            __builtin_prefetch((const void*)(Vt + ((size_t)b * CDIM + ch * 128 + lm) * NPIX + j0 + 32), 0, 0);
        }
        // S = Q K^T for 32 keys (two 16x16 tiles)
        v8f s0 = (v8f){0,0,0,0,0,0,0,0};
        v8f s1 = (v8f){0,0,0,0,0,0,0,0};
        #pragma unroll
        for (int kc = 0; kc < 4; ++kc) {
            H16 k0u; k0u.u[0] = *(const uint4*)(Km + ((size_t)b * NPIX + j0 + lm) * DDIM + kc * 32 + abase); k0u.u[1] = *(const uint4*)(Km + ((size_t)b * NPIX + j0 + lm) * DDIM + kc * 32 + abase + 16); v16h bk0 = k0u.v;
            s0 = wmma16(qf[kc].v, bk0, s0);
            H16 k1u; k1u.u[0] = *(const uint4*)(Km + ((size_t)b * NPIX + j0 + 16 + lm) * DDIM + kc * 32 + abase); k1u.u[1] = *(const uint4*)(Km + ((size_t)b * NPIX + j0 + 16 + lm) * DDIM + kc * 32 + abase + 16); v16h bk1 = k1u.v;
            s1 = wmma16(qf[kc].v, bk1, s1);
        }
        // Online softmax update; write P (f16) to wave-private LDS.
        #pragma unroll
        for (int r = 0; r < 8; ++r) {
            float a0 = s0[r] * scale, a1 = s1[r] * scale;
            float bm = redmax16(fmaxf(a0, a1));
            float nm = fmaxf(rmax[r], bm);
            float f  = __expf(rmax[r] - nm);
            rmax[r]  = nm;
            float p0 = __expf(a0 - nm);
            float p1 = __expf(a1 - nm);
            rsum[r]  = rsum[r] * f + redsum16(p0 + p1);
            #pragma unroll
            for (int ot = 0; ot < 8; ++ot) acc[ot][r] *= f;
            int m = r + (lh ? 8 : 0);
            pbuf[w][lm * 16 + m]        = (_Float16)p0;  // key j0+lm
            pbuf[w][(16 + lm) * 16 + m] = (_Float16)p1;  // key j0+16+lm
        }
        // P A-fragment (transpose via LDS), then O += P * V
        v16h pa;
        #pragma unroll
        for (int h = 0; h < 16; ++h) {
            int k = (h < 8) ? (abase + h) : (abase + 16 + (h - 8));
            pa[h] = pbuf[w][k * 16 + lm];
        }
        #pragma unroll
        for (int ot = 0; ot < 8; ++ot) {
            int o = ch * 128 + ot * 16 + lm;
            H16 vu; vu.u[0] = *(const uint4*)(Vt + ((size_t)b * CDIM + o) * NPIX + j0 + abase); vu.u[1] = *(const uint4*)(Vt + ((size_t)b * CDIM + o) * NPIX + j0 + abase + 16); v16h bv = vu.v;
            acc[ot] = wmma16(pa, bv, acc[ot]);
        }
    }

    // Normalize and store (row-major N x C, f16)
    #pragma unroll
    for (int r = 0; r < 8; ++r) {
        float inv = 1.0f / rsum[r];
        int n = n0 + r + (lh ? 8 : 0);
        #pragma unroll
        for (int ot = 0; ot < 8; ++ot) {
            int o = ch * 128 + ot * 16 + lm;
            Out[((size_t)b * NPIX + n) * CDIM + o] = (_Float16)(acc[ot][r] * inv);
        }
    }
}

// ---------------------------------------------------------------------------
// Fuse: z = relu(Xs*W1^T + Xa*W2^T + bias) + resid   (W = fuse_w (C x 2C))
// Xs f32 channel-major (B,C,N); Xa f16 row-major (B,N,C); z f32 (B,C,N).
// Also accumulates per-batch sum / sumsq for the LayerNorm.
// Block = 8 waves: 4 row-tiles x 2 channel-halves; LDS slice shared per pair.
// ---------------------------------------------------------------------------
__global__ __launch_bounds__(256) void fuse_kernel(const float* __restrict__ Xs,
                                                   const _Float16* __restrict__ Xa,
                                                   const _Float16* __restrict__ Wf,
                                                   const float* __restrict__ bias,
                                                   const float* __restrict__ resid,
                                                   float* __restrict__ Z,
                                                   float* __restrict__ stats) {
    __shared__ _Float16 lds[4][16 * CDIM];
    const int w    = threadIdx.x >> 5;
    const int lane = threadIdx.x & 31;
    const int lm   = lane & 15;
    const int lh   = lane >> 4;
    const int rt   = w >> 1;                 // row tile (0..3)
    const int ch   = w & 1;                  // channel half
    const int b    = blockIdx.x >> 4;        // 16 blocks per batch
    const int n0   = (blockIdx.x & 15) * 64 + rt * 16;
    const int abase = lh ? 8 : 0;
    const int kbB   = lh ? 16 : 0;

    // Cooperative staging of Xs tile: wave stages its channel half.
    #pragma unroll
    for (int cc = 0; cc < 4; ++cc) {
        int c = ch * 128 + cc * 32 + lane;
        const float* src = Xs + ((size_t)b * CDIM + c) * NPIX + n0;
        #pragma unroll
        for (int i = 0; i < 16; i += 4) {
            float4 f = *(const float4*)(src + i);
            lds[rt][c * 16 + i + 0] = (_Float16)f.x;
            lds[rt][c * 16 + i + 1] = (_Float16)f.y;
            lds[rt][c * 16 + i + 2] = (_Float16)f.z;
            lds[rt][c * 16 + i + 3] = (_Float16)f.w;
        }
    }
    __syncthreads();

    v8f acc[8];
    #pragma unroll
    for (int ot = 0; ot < 8; ++ot) acc[ot] = (v8f){0,0,0,0,0,0,0,0};

    const _Float16* arow = Xa + ((size_t)b * NPIX + n0 + lm) * CDIM;
    #pragma unroll
    for (int kc = 0; kc < 8; ++kc) {
        v16h a1;                               // from Xs (W columns 0..255)
        #pragma unroll
        for (int h = 0; h < 16; ++h) {
            int k = kc * 32 + ((h < 8) ? (abase + h) : (abase + 16 + (h - 8)));
            a1[h] = lds[rt][k * 16 + lm];
        }
        H16 a2;                                // from Xa (W columns 256..511)
        a2.u[0] = *(const uint4*)(arow + kc * 32 + abase);
        a2.u[1] = *(const uint4*)(arow + kc * 32 + abase + 16);

        #pragma unroll
        for (int ot = 0; ot < 8; ++ot) {
            int o = ch * 128 + ot * 16 + lm;
            H16 b1u; b1u.u[0] = *(const uint4*)(Wf + (size_t)o * 512 + kc * 32 + abase); b1u.u[1] = *(const uint4*)(Wf + (size_t)o * 512 + kc * 32 + abase + 16); v16h b1 = b1u.v;
            H16 b2u; b2u.u[0] = *(const uint4*)(Wf + (size_t)o * 512 + 256 + kc * 32 + abase); b2u.u[1] = *(const uint4*)(Wf + (size_t)o * 512 + 256 + kc * 32 + abase + 16); v16h b2 = b2u.v;
            acc[ot] = wmma16(a1,   b1, acc[ot]);
            acc[ot] = wmma16(a2.v, b2, acc[ot]);
        }
    }

    float s1 = 0.0f, s2 = 0.0f;
    #pragma unroll
    for (int ot = 0; ot < 8; ++ot) {
        int o = ch * 128 + ot * 16 + lm;
        float bf = bias[o];
        const float* rrow = resid + ((size_t)b * CDIM + o) * NPIX + n0 + (lh ? 8 : 0);
        float*       zrow = Z     + ((size_t)b * CDIM + o) * NPIX + n0 + (lh ? 8 : 0);
        #pragma unroll
        for (int r = 0; r < 8; ++r) {
            float z = fmaxf(acc[ot][r] + bf, 0.0f) + rrow[r];
            zrow[r] = z;
            s1 += z;
            s2 += z * z;
        }
    }
    s1 = redsum32(s1);
    s2 = redsum32(s2);
    if (lane == 0) {
        atomicAdd(&stats[b * 2 + 0], s1);
        atomicAdd(&stats[b * 2 + 1], s2);
    }
}

// ---------------------------------------------------------------------------
// LayerNorm apply (in place on d_out), both paths.
// ---------------------------------------------------------------------------
__global__ __launch_bounds__(256) void ln_kernel(float* __restrict__ out,
                                                 const float* __restrict__ stats,
                                                 const float* __restrict__ ws_w,
                                                 const float* __restrict__ ws_b,
                                                 const float* __restrict__ wi_w,
                                                 const float* __restrict__ wi_b) {
    const size_t PER = (size_t)BATCH * CDIM * NPIX;     // per path
    size_t base = ((size_t)blockIdx.x * 256 + threadIdx.x) * 4;
    #pragma unroll
    for (int i = 0; i < 4; ++i) {
        size_t id  = base + i;
        int path   = id >= PER;
        size_t rem = id - (size_t)path * PER;
        int b = (int)(rem >> 18);                       // / (256*1024)
        int c = (int)(rem >> 10) & 255;
        const float* st = stats + ((size_t)path * 32 + b) * 2;
        float mean = st[0] * (1.0f / 262144.0f);
        float var  = st[1] * (1.0f / 262144.0f) - mean * mean;
        float rstd = rsqrtf(var + 1e-5f);
        float wv = path ? wi_w[c] : ws_w[c];
        float bv = path ? wi_b[c] : ws_b[c];
        out[id] = (out[id] - mean) * rstd * wv + bv;
    }
}

// ---------------------------------------------------------------------------
// Host launcher
// ---------------------------------------------------------------------------
extern "C" void kernel_launch(void* const* d_in, const int* in_sizes, int n_in,
                              void* d_out, int out_size, void* d_ws, size_t ws_size,
                              hipStream_t stream) {
    (void)in_sizes; (void)n_in; (void)out_size; (void)ws_size;

    const float* fs     = (const float*)d_in[0];
    const float* fi     = (const float*)d_in[1];
    const float* qs_w   = (const float*)d_in[2];
    const float* ks_w   = (const float*)d_in[3];
    const float* vs_w   = (const float*)d_in[4];
    const float* qi_w   = (const float*)d_in[5];
    const float* ki_w   = (const float*)d_in[6];
    const float* vi_w   = (const float*)d_in[7];
    const float* fuse_w = (const float*)d_in[8];
    const float* fuse_b = (const float*)d_in[9];
    const float* ln_s_w = (const float*)d_in[10];
    const float* ln_s_b = (const float*)d_in[11];
    const float* ln_i_w = (const float*)d_in[12];
    const float* ln_i_b = (const float*)d_in[13];

    char* ws = (char*)d_ws;
    size_t off = 0;
    auto take = [&](size_t bytes) -> char* {
        char* p = ws + off;
        off += (bytes + 255) & ~(size_t)255;
        return p;
    };

    _Float16* wh_qs   = (_Float16*)take((size_t)DDIM * CDIM * 2);
    _Float16* wh_ks   = (_Float16*)take((size_t)DDIM * CDIM * 2);
    _Float16* wh_vs   = (_Float16*)take((size_t)CDIM * CDIM * 2);
    _Float16* wh_qi   = (_Float16*)take((size_t)DDIM * CDIM * 2);
    _Float16* wh_ki   = (_Float16*)take((size_t)DDIM * CDIM * 2);
    _Float16* wh_vi   = (_Float16*)take((size_t)CDIM * CDIM * 2);
    _Float16* wh_fuse = (_Float16*)take((size_t)CDIM * 512 * 2);
    float*    stats   = (float*)take(128 * 4);
    _Float16* Qs      = (_Float16*)take((size_t)BATCH * NPIX * DDIM * 2);
    _Float16* Ks      = (_Float16*)take((size_t)BATCH * NPIX * DDIM * 2);
    _Float16* Qi      = (_Float16*)take((size_t)BATCH * NPIX * DDIM * 2);
    _Float16* Ki      = (_Float16*)take((size_t)BATCH * NPIX * DDIM * 2);
    _Float16* VsT     = (_Float16*)take((size_t)BATCH * CDIM * NPIX * 2);
    _Float16* ViT     = (_Float16*)take((size_t)BATCH * CDIM * NPIX * 2);
    _Float16* s2i     = (_Float16*)take((size_t)BATCH * NPIX * CDIM * 2);
    _Float16* i2s     = (_Float16*)take((size_t)BATCH * NPIX * CDIM * 2);

    // 0) zero LN stats
    zero_kernel<<<1, 128, 0, stream>>>(stats, 128);

    // 1) weights -> f16
    cvt_kernel<<<(DDIM*CDIM + 255) / 256, 256, 0, stream>>>(qs_w, wh_qs, DDIM*CDIM);
    cvt_kernel<<<(DDIM*CDIM + 255) / 256, 256, 0, stream>>>(ks_w, wh_ks, DDIM*CDIM);
    cvt_kernel<<<(CDIM*CDIM + 255) / 256, 256, 0, stream>>>(vs_w, wh_vs, CDIM*CDIM);
    cvt_kernel<<<(DDIM*CDIM + 255) / 256, 256, 0, stream>>>(qi_w, wh_qi, DDIM*CDIM);
    cvt_kernel<<<(DDIM*CDIM + 255) / 256, 256, 0, stream>>>(ki_w, wh_ki, DDIM*CDIM);
    cvt_kernel<<<(CDIM*CDIM + 255) / 256, 256, 0, stream>>>(vi_w, wh_vi, CDIM*CDIM);
    cvt_kernel<<<(CDIM*512  + 255) / 256, 256, 0, stream>>>(fuse_w, wh_fuse, CDIM*512);

    // 2) projections (grid: 16 blocks/batch x 32 batches, 4 waves/block)
    proj_kernel<DDIM, false><<<512, 128, 0, stream>>>(fs, wh_qs, Qs);
    proj_kernel<DDIM, false><<<512, 128, 0, stream>>>(fs, wh_ks, Ks);
    proj_kernel<CDIM, true ><<<512, 128, 0, stream>>>(fs, wh_vs, VsT);
    proj_kernel<DDIM, false><<<512, 128, 0, stream>>>(fi, wh_qi, Qi);
    proj_kernel<DDIM, false><<<512, 128, 0, stream>>>(fi, wh_ki, Ki);
    proj_kernel<CDIM, true ><<<512, 128, 0, stream>>>(fi, wh_vi, ViT);

    // 3) cross attention
    attn_kernel<<<1024, 128, 0, stream>>>(Qs, Ki, ViT, s2i);   // sketch q over image k/v
    attn_kernel<<<1024, 128, 0, stream>>>(Qi, Ks, VsT, i2s);   // image q over sketch k/v

    // 4) fuse + residual, write pre-LN z into d_out, accumulate LN stats
    float* zs = (float*)d_out;
    float* zi = zs + (size_t)BATCH * CDIM * NPIX;
    fuse_kernel<<<512, 256, 0, stream>>>(fs, i2s, wh_fuse, fuse_b, fs, zs, stats + 0);
    fuse_kernel<<<512, 256, 0, stream>>>(fi, s2i, wh_fuse, fuse_b, fi, zi, stats + 64);

    // 5) LayerNorm apply in place on d_out (both paths)
    ln_kernel<<<16384, 256, 0, stream>>>((float*)d_out, stats,
                                         ln_s_w, ln_s_b, ln_i_w, ln_i_b);
}
